// S2VT_31756988187133
// MI455X (gfx1250) — matrix operand order repaired and checked by reading, not compiled, not run on hardware
//
#include <hip/hip_runtime.h>
#include <hip/hip_bf16.h>

// ---------------- problem constants ----------------
#define B_    128
#define T_    80
#define FEAT_ 4096
#define H_    512
#define H4_   2048
#define OHL_  3000
#define ML_   44
#define KWI_  3008   // 3000 padded up to multiple of 32

typedef __attribute__((ext_vector_type(16))) __bf16 v16bf;
typedef __attribute__((ext_vector_type(8)))  float  v8f;

struct U8 { unsigned u[8]; };

__device__ __forceinline__ unsigned short f2bf(float f) {
  unsigned u = __builtin_bit_cast(unsigned, f);
  u += 0x7FFFu + ((u >> 16) & 1u);          // round-to-nearest-even
  return (unsigned short)(u >> 16);
}

// A-matrix 16x32 bf16 VGPR layout (per lane, lane-half g = lane>>4):
// elements 0..15 -> k = kbase + g*16 + j (matches B fragment)
// With bf16-resident A these are two contiguous 16B runs -> 2 x b128 loads.
__device__ __forceinline__ void load_abf_frag(const unsigned short* __restrict__ Arow,
                                              int kbase, int g, U8& out) {
  const uint4 a = *(const uint4*)(Arow + kbase + g * 16);
  const uint4 b = *(const uint4*)(Arow + kbase + g * 16 + 8);
  out.u[0]=a.x; out.u[1]=a.y; out.u[2]=a.z; out.u[3]=a.w;
  out.u[4]=b.x; out.u[5]=b.y; out.u[6]=b.z; out.u[7]=b.w;
}

// B-matrix 32x16 bf16 VGPR layout: lane-group g holds k = kbase + g*16 + j (j=0..15),
// column n = lane&15. Contiguous 32B per lane -> 2 x b128 loads.
__device__ __forceinline__ void load_bbf_frag(const unsigned short* __restrict__ Brow,
                                              int kbase, int g, U8& out) {
  const unsigned* q = (const unsigned*)(Brow + kbase + g * 16);
  const uint4 a = *(const uint4*)(q);
  const uint4 b = *(const uint4*)(q + 4);
  out.u[0]=a.x; out.u[1]=a.y; out.u[2]=a.z; out.u[3]=a.w;
  out.u[4]=b.x; out.u[5]=b.y; out.u[6]=b.z; out.u[7]=b.w;
}

// C = A(bf16, MxKpad, lda) x Bw(bf16, NxKpad row-major weights)^T (+ bias) (+ Cadd)
// Block: 128 threads = 4 waves; wave tile 32(M) x 64(N): 2 A-frags x 4 B-frags
// -> 8 v_wmma_f32_16x16x32_bf16 per 32-wide K step against 12 b128 loads, ~no VALU.
// Requires: M % 128 == 0, Kpad % 32 == 0, lda % 8 == 0 (all true for every call).
__global__ __launch_bounds__(128)
void gemm_wmma(const unsigned short* __restrict__ A, int lda, int Kpad,
               const unsigned short* __restrict__ Bw,
               void* __restrict__ Cv, int ldc, int storeBf16,
               const float* __restrict__ Cadd, int ldadd,
               const float* __restrict__ bias, int N)
{
  const int lane = threadIdx.x & 31;
  const int wave = threadIdx.x >> 5;
  const int g    = lane >> 4;
  const int l16  = lane & 15;
  const int nb   = blockIdx.x * 64;
  const int mw   = blockIdx.y * 128 + wave * 32;

  const unsigned short* Arow0 = A + (size_t)(mw + l16) * lda;
  const unsigned short* Arow1 = Arow0 + (size_t)16 * lda;

  const unsigned short* Brow[4];
#pragma unroll
  for (int j = 0; j < 4; ++j) {
    int n = nb + j * 16 + l16;
    if (n > N - 1) n = N - 1;               // clamp reads; stores are masked
    Brow[j] = Bw + (size_t)n * Kpad;
  }

  v8f acc[2][4];
  const v8f zacc = {0.f, 0.f, 0.f, 0.f, 0.f, 0.f, 0.f, 0.f};
#pragma unroll
  for (int i = 0; i < 2; ++i)
#pragma unroll
    for (int j = 0; j < 4; ++j) acc[i][j] = zacc;

  for (int kb = 0; kb < Kpad; kb += 32) {
    U8 au0, au1, bu0, bu1, bu2, bu3;
    load_abf_frag(Arow0, kb, g, au0);
    load_abf_frag(Arow1, kb, g, au1);
    load_bbf_frag(Brow[0], kb, g, bu0);
    load_bbf_frag(Brow[1], kb, g, bu1);
    load_bbf_frag(Brow[2], kb, g, bu2);
    load_bbf_frag(Brow[3], kb, g, bu3);
    v16bf a0 = __builtin_bit_cast(v16bf, au0);
    v16bf a1 = __builtin_bit_cast(v16bf, au1);
    v16bf b0 = __builtin_bit_cast(v16bf, bu0);
    v16bf b1 = __builtin_bit_cast(v16bf, bu1);
    v16bf b2 = __builtin_bit_cast(v16bf, bu2);
    v16bf b3 = __builtin_bit_cast(v16bf, bu3);
    acc[0][0] = __builtin_amdgcn_wmma_f32_16x16x32_bf16(false, a0, false, b0, (short)0, acc[0][0], false, false);
    acc[0][1] = __builtin_amdgcn_wmma_f32_16x16x32_bf16(false, a0, false, b1, (short)0, acc[0][1], false, false);
    acc[0][2] = __builtin_amdgcn_wmma_f32_16x16x32_bf16(false, a0, false, b2, (short)0, acc[0][2], false, false);
    acc[0][3] = __builtin_amdgcn_wmma_f32_16x16x32_bf16(false, a0, false, b3, (short)0, acc[0][3], false, false);
    acc[1][0] = __builtin_amdgcn_wmma_f32_16x16x32_bf16(false, a1, false, b0, (short)0, acc[1][0], false, false);
    acc[1][1] = __builtin_amdgcn_wmma_f32_16x16x32_bf16(false, a1, false, b1, (short)0, acc[1][1], false, false);
    acc[1][2] = __builtin_amdgcn_wmma_f32_16x16x32_bf16(false, a1, false, b2, (short)0, acc[1][2], false, false);
    acc[1][3] = __builtin_amdgcn_wmma_f32_16x16x32_bf16(false, a1, false, b3, (short)0, acc[1][3], false, false);
  }

  // C/D layout: VGPR r -> row (r + 8*g), col = lane&15 within each 16x16 tile.
#pragma unroll
  for (int j = 0; j < 4; ++j) {
    const int col = nb + j * 16 + l16;
    if (col < N) {
      const float bv = bias ? bias[col] : 0.f;
#pragma unroll
      for (int i = 0; i < 2; ++i) {
        const int mbase = mw + i * 16 + g * 8;
#pragma unroll
        for (int r = 0; r < 8; ++r) {
          float v = acc[i][j][r] + bv;
          const size_t row = (size_t)(mbase + r);
          if (Cadd) v += Cadd[row * (size_t)ldadd + col];
          if (storeBf16) ((unsigned short*)Cv)[row * (size_t)ldc + col] = f2bf(v);
          else           ((float*)Cv)[row * (size_t)ldc + col] = v;
        }
      }
    }
  }
}

// fp32 (N x Kact) -> bf16 (N x Kpad), zero-padded tail (weights AND activations)
__global__ void cvt_bf(const float* __restrict__ src, unsigned short* __restrict__ dst,
                       int Kact, int Kpad, long long total) {
  long long i = (long long)blockIdx.x * blockDim.x + threadIdx.x;
  if (i >= total) return;
  long long n = i / Kpad;
  int k = (int)(i - n * Kpad);
  float v = (k < Kact) ? src[n * (long long)Kact + k] : 0.f;
  dst[i] = f2bf(v);
}

__global__ void vec_add(const float* __restrict__ a, const float* __restrict__ b,
                        float* __restrict__ o, int n) {
  int i = blockIdx.x * blockDim.x + threadIdx.x;
  if (i < n) o[i] = a[i] + b[i];
}

__global__ void fill_zero(float* __restrict__ p, long long n) {
  long long i = (long long)blockIdx.x * blockDim.x + threadIdx.x;
  if (i < n) p[i] = 0.f;
}

// gates (B x 4H) pre-summed (xW + hW + bias); PyTorch order i,f,g,o.
// c stays fp32; h is emitted as bf16 (only ever consumed as a GEMM A operand).
__global__ void lstm_cell(const float* __restrict__ gates,
                          float* __restrict__ c, unsigned short* __restrict__ hbf,
                          unsigned short* __restrict__ aux, int auxStride) {
  int idx = blockIdx.x * blockDim.x + threadIdx.x;
  if (idx >= B_ * H_) return;
  int b = idx >> 9;           // / 512
  int d = idx & (H_ - 1);
  const float* gr = gates + (size_t)b * H4_;
  float xi = gr[d];
  float xf = gr[H_ + d];
  float xg = gr[2 * H_ + d];
  float xo = gr[3 * H_ + d];
  float ig = 1.f / (1.f + __expf(-xi));
  float fg = 1.f / (1.f + __expf(-xf));
  float gg = tanhf(xg);
  float og = 1.f / (1.f + __expf(-xo));
  float cv = fg * c[idx] + ig * gg;
  float hv = og * tanhf(cv);
  c[idx] = cv;
  unsigned short hb = f2bf(hv);
  hbf[idx] = hb;
  if (aux) aux[(size_t)b * auxStride + d] = hb;
}

// samples[0] = one-hot(OHL-2) @ Wi^T + bi, broadcast over batch -> dec_in2[0,:,0:512] (bf16)
__global__ void bos_embed(const float* __restrict__ Wi, const float* __restrict__ bi,
                          unsigned short* __restrict__ dst) {
  int idx = blockIdx.x * blockDim.x + threadIdx.x;
  if (idx >= B_ * H_) return;
  int b = idx >> 9;
  int d = idx & (H_ - 1);
  dst[(size_t)b * 1024 + d] = f2bf(Wi[(size_t)d * OHL_ + (OHL_ - 2)] + bi[d]);
}

extern "C" void kernel_launch(void* const* d_in, const int* in_sizes, int n_in,
                              void* d_out, int out_size, void* d_ws, size_t ws_size,
                              hipStream_t stream) {
  (void)in_sizes; (void)n_in; (void)out_size; (void)ws_size;
  const float* x     = (const float*)d_in[0];   // (128,80,4096) == flat (80*128,4096)
  const float* ans   = (const float*)d_in[1];   // (44,128,3000)
  const float* We_ih = (const float*)d_in[2];
  const float* We_hh = (const float*)d_in[3];
  const float* be_ih = (const float*)d_in[4];
  const float* be_hh = (const float*)d_in[5];
  const float* Wd_ih = (const float*)d_in[6];
  const float* Wd_hh = (const float*)d_in[7];
  const float* bd_ih = (const float*)d_in[8];
  const float* bd_hh = (const float*)d_in[9];
  const float* Wi    = (const float*)d_in[10];
  const float* bi    = (const float*)d_in[11];
  const float* Wo    = (const float*)d_in[12];
  const float* bo    = (const float*)d_in[13];
  float* out = (float*)d_out;

  // ---- workspace carve-out (256B aligned) ----
  char* ws = (char*)d_ws;
  size_t off = 0;
  auto alloc = [&](size_t bytes) -> void* {
    void* p = ws + off;
    off = (off + bytes + 255) & ~(size_t)255;
    return p;
  };
  unsigned short* We_ih_bf = (unsigned short*)alloc((size_t)H4_ * FEAT_ * 2);
  unsigned short* We_hh_bf = (unsigned short*)alloc((size_t)H4_ * H_ * 2);
  unsigned short* Wd_ih_bf = (unsigned short*)alloc((size_t)H4_ * 1024 * 2);
  unsigned short* Wd_hh_bf = (unsigned short*)alloc((size_t)H4_ * H_ * 2);
  unsigned short* Wi_bf    = (unsigned short*)alloc((size_t)H_ * KWI_ * 2);
  unsigned short* Wo_bf    = (unsigned short*)alloc((size_t)OHL_ * H_ * 2);
  float* be   = (float*)alloc((size_t)H4_ * 4);
  float* bd   = (float*)alloc((size_t)H4_ * 4);
  float* Gbig = (float*)alloc((size_t)T_ * B_ * H4_ * 4);     // batched input projections

  // Aliased region: x_bf (84MB) is dead after the first big GEMM; the caption-phase
  // bf16 buffers (71MB total) live in the same storage afterwards.
  char* regionX = (char*)alloc((size_t)T_ * B_ * FEAT_ * 2);
  unsigned short* x_bf    = (unsigned short*)regionX;
  unsigned short* dec_in1 = (unsigned short*)regionX;                                   // 80*128*1024 bf16
  unsigned short* ans_bf  = (unsigned short*)(regionX + (size_t)T_ * B_ * 1024 * 2);    // 5504*3008 bf16
  unsigned short* dec_in2 = (unsigned short*)(regionX + (size_t)T_ * B_ * 1024 * 2
                                                      + (size_t)(ML_ - 1) * B_ * KWI_ * 2);
  unsigned short* dec_out = (unsigned short*)((char*)dec_in2 + (size_t)ML_ * B_ * 1024 * 2);

  float* gates = (float*)alloc((size_t)B_ * H4_ * 4);
  unsigned short* he = (unsigned short*)alloc((size_t)B_ * H_ * 2);
  unsigned short* hd = (unsigned short*)alloc((size_t)B_ * H_ * 2);
  float* ce = (float*)alloc((size_t)B_ * H_ * 4);
  float* cd = (float*)alloc((size_t)B_ * H_ * 4);

  auto cvt = [&](const float* s, unsigned short* d, long long rows, int kact, int kpad) {
    long long tot = rows * kpad;
    cvt_bf<<<dim3((unsigned)((tot + 255) / 256)), 256, 0, stream>>>(s, d, kact, kpad, tot);
  };
  auto zero = [&](void* p, long long nbytes) {  // nbytes % 4 == 0 for all calls
    fill_zero<<<dim3((unsigned)((nbytes / 4 + 255) / 256)), 256, 0, stream>>>((float*)p, nbytes / 4);
  };
  auto gemm = [&](const unsigned short* A, int lda, int Kpad, const unsigned short* Bw,
                  void* C, int ldc, int storeBf16, const float* Cadd, int ldadd,
                  const float* bias, int M, int N) {
    dim3 grid((unsigned)((N + 63) / 64), (unsigned)(M / 128));
    gemm_wmma<<<grid, 128, 0, stream>>>(A, lda, Kpad, Bw, C, ldc, storeBf16,
                                        Cadd, ldadd, bias, N);
  };
  auto cell = [&](float* c, unsigned short* h, unsigned short* aux, int auxStride) {
    lstm_cell<<<(B_ * H_ + 255) / 256, 256, 0, stream>>>(gates, c, h, aux, auxStride);
  };

  // ---- one-time prep ----
  cvt(We_ih, We_ih_bf, H4_, FEAT_, FEAT_);
  cvt(We_hh, We_hh_bf, H4_, H_, H_);
  cvt(Wd_ih, Wd_ih_bf, H4_, 1024, 1024);
  cvt(Wd_hh, Wd_hh_bf, H4_, H_, H_);
  cvt(Wi,    Wi_bf,    H_, OHL_, KWI_);     // K zero-padded 3000 -> 3008
  cvt(Wo,    Wo_bf,    OHL_, H_, H_);
  vec_add<<<(H4_ + 255) / 256, 256, 0, stream>>>(be_ih, be_hh, be, H4_);
  vec_add<<<(H4_ + 255) / 256, 256, 0, stream>>>(bd_ih, bd_hh, bd, H4_);
  zero(he, (size_t)B_ * H_ * 2);
  zero(hd, (size_t)B_ * H_ * 2);
  zero(ce, (size_t)B_ * H_ * 4);
  zero(cd, (size_t)B_ * H_ * 4);

  // ---- phase 1: encoder input projection for all 80 steps (the big GEMM) ----
  cvt(x, x_bf, (long long)T_ * B_, FEAT_, FEAT_);
  gemm(x_bf, FEAT_, FEAT_, We_ih_bf, Gbig, H4_, 0, nullptr, 0, be, T_ * B_, H4_);

  // x_bf is now dead -> regionX is reused for the bf16 sequence buffers.
  zero(dec_in1, (size_t)T_ * B_ * 1024 * 2);   // second 512 cols stay zero

  // ---- encoder recurrence; h_t also written into dec_in1[t,:,0:512] (bf16) ----
  for (int t = 0; t < T_; ++t) {
    gemm(he, H_, H_, We_hh_bf, gates, H4_, 0, Gbig + (size_t)t * B_ * H4_, H4_,
         nullptr, B_, H4_);
    cell(ce, he, dec_in1 + (size_t)t * B_ * 1024, 1024);
  }

  // ---- decoder pass 1: batched input projection, then recurrence ----
  gemm(dec_in1, 1024, 1024, Wd_ih_bf, Gbig, H4_, 0, nullptr, 0, bd, T_ * B_, H4_);
  for (int t = 0; t < T_; ++t) {
    gemm(hd, H_, H_, Wd_hh_bf, gates, H4_, 0, Gbig + (size_t)t * B_ * H4_, H4_,
         nullptr, B_, H4_);
    cell(cd, hd, nullptr, 0);
  }

  // ---- encoder over zero padding (x part is exactly zero -> just h@Whh + be) ----
  for (int t = 0; t < ML_; ++t) {
    gemm(he, H_, H_, We_hh_bf, gates, H4_, 0, nullptr, 0, be, B_, H4_);
    cell(ce, he, dec_in2 + (size_t)t * B_ * 1024 + H_, 1024);  // cols 512..1023
  }

  // ---- caption inputs: BOS + teacher-forced embeddings into dec_in2[:,:,0:512] (bf16) ----
  bos_embed<<<(B_ * H_ + 255) / 256, 256, 0, stream>>>(Wi, bi, dec_in2);
  cvt(ans + (size_t)B_ * OHL_, ans_bf, (long long)(ML_ - 1) * B_, OHL_, KWI_);
  gemm(ans_bf, KWI_, KWI_, Wi_bf, dec_in2 + (size_t)B_ * 1024, 1024, 1,
       nullptr, 0, bi, (ML_ - 1) * B_, H_);

  // ---- decoder pass 2: batched input projection, then recurrence ----
  gemm(dec_in2, 1024, 1024, Wd_ih_bf, Gbig, H4_, 0, nullptr, 0, bd, ML_ * B_, H4_);
  for (int t = 0; t < ML_; ++t) {
    gemm(hd, H_, H_, Wd_hh_bf, gates, H4_, 0, Gbig + (size_t)t * B_ * H4_, H4_,
         nullptr, B_, H4_);
    cell(cd, hd, dec_out + (size_t)t * B_ * H_, H_);
  }

  // ---- final projection to vocab (N=3000 handled by clamp/mask) ----
  gemm(dec_out, H_, H_, Wo_bf, out, OHL_, 0, nullptr, 0, bo, ML_ * B_, OHL_);
}
